// ModNN_1666447311083
// MI455X (gfx1250) — matrix operand order repaired and checked by reading, not compiled, not run on hardware
//
#include <hip/hip_runtime.h>

typedef __attribute__((ext_vector_type(16))) _Float16 v16h;
typedef __attribute__((ext_vector_type(8)))  float    v8f;

#define B_     512
#define T_     256
#define HID_   128
#define GATES_ 512
#define WIN_   8
#define ENCO_  128
#define BT_    (B_*T_)

#define LOG2E  1.44269504088896340736f

// branch-free fast activations on the hardware transcendental pipe
__device__ __forceinline__ float sigf(float x){
  // 1 / (1 + exp(-x)) : v_exp_f32 + v_add + v_rcp_f32, no branches
  float t = __builtin_amdgcn_exp2f(-LOG2E * x);
  return __builtin_amdgcn_rcpf(1.0f + t);
}
__device__ __forceinline__ float tanhfast(float x){
  // tanh(x) = sign(x) * (1 - t) / (1 + t),  t = exp(-2|x|)
  float ax = __builtin_fabsf(x);
  float t  = __builtin_amdgcn_exp2f(-2.0f * LOG2E * ax);
  float r  = (1.0f - t) * __builtin_amdgcn_rcpf(1.0f + t);
  return __builtin_copysignf(r, x);
}

// ---------------- prep: pack W_hh (512x128 f32) into f16 B-fragment order ----------------
// Gate-permuted tiling: flat tile nt = wave*4 + q, where tile q holds GATE q
// (i,f,g,o) for hidden units j = wave*16 .. wave*16+15.  So the original W_hh
// row for tile nt, local column n is:  n_glob = (nt&3)*128 + (nt>>2)*16 + n.
// B operand (K x N = 32 x 16) lane layout: lane l: n = l&15, K = (l>>4)*16 + e.
// flat f16 index = ((nt*4 + kstep)*32 + lane)*16 + e
__global__ void pack_whh(const float* __restrict__ Whh, _Float16* __restrict__ out){
  int i = blockIdx.x*blockDim.x + threadIdx.x;      // 65536 elements
  if(i >= GATES_*HID_) return;
  int e = i & 15, lane = (i>>4)&31, ks = (i>>9)&3, nt = i>>11;
  int n = (nt&3)*128 + (nt>>2)*16 + (lane&15);      // gate-permuted row of W_hh
  int k = ks*32 + ((lane>>4)<<4) + e;
  out[i] = (_Float16)Whh[n*HID_ + k];
}

__global__ void pack_fc1(const float* __restrict__ Wfc1, _Float16* __restrict__ out){
  int i = blockIdx.x*blockDim.x + threadIdx.x;      // 16384 elements
  if(i >= HID_*HID_) return;
  int e = i & 15, lane = (i>>4)&31, ks = (i>>9)&3, nt = i>>11;
  int n = nt*16 + (lane&15);
  int k = ks*32 + ((lane>>4)<<4) + e;
  out[i] = (_Float16)Wfc1[n*HID_ + k];
}

// ---------------- int_all precompute ----------------
__global__ void int_all_kernel(const float* __restrict__ X,
                               const float* __restrict__ w1, const float* __restrict__ b1,
                               const float* __restrict__ w3, const float* __restrict__ b3,
                               const float* __restrict__ sw, float* __restrict__ out){
  int i = blockIdx.x*blockDim.x + threadIdx.x;      // B*T
  if(i >= BT_) return;
  const float* xp = X + i*7 + 3;                    // channels 3,4,5
  float x0 = xp[0], x1 = xp[1], x2 = xp[2];
  float acc = 0.f;
  #pragma unroll 4
  for(int h=0; h<HID_; ++h){
    float v = fmaf(w1[h*3+0], x0, fmaf(w1[h*3+1], x1, fmaf(w1[h*3+2], x2, b1[h])));
    v = fmaxf(v, 0.f);
    acc = fmaf(v, w3[h], acc);
  }
  out[i] = sigf(acc + b3[0]) * sw[0];
}

// ---------------- main sequential kernel: 1 block = 16 batch rows, 8 waves ----------------
__global__ __launch_bounds__(256)
void modnn_main(const float* __restrict__ X,
                const float* __restrict__ Wih,
                const float* __restrict__ bih, const float* __restrict__ bhh,
                const float* __restrict__ fc1b,
                const float* __restrict__ fc2w, const float* __restrict__ fc2b,
                const float* __restrict__ zonew,
                const _Float16* __restrict__ whhF, const _Float16* __restrict__ fc1F,
                const float* __restrict__ intall,
                float* __restrict__ out)
{
  __shared__ float    sTout[16*T_];        // 16 KB : this block's TOut rows
  __shared__ float    sO[16*HID_];         //  8 KB : fc1 output
  __shared__ _Float16 sHf[2][4*32*16];     //  8 KB : h in A-fragment order, double-buffered
  __shared__ float    sX[16*WIN_*5];       // 2.5KB : embed window
  __shared__ float    sE[16];
  __shared__ float    sRed[256];

  const int tid  = threadIdx.x;
  const int lane = tid & 31;
  const int wv   = tid >> 5;               // 0..7
  const int rowBase = blockIdx.x * 16;
  const int m    = lane & 15;              // C/D column lane, A row
  const int hi8  = (lane >> 4) << 3;       // +8 row offset for C/D rows in lanes 16..31

  // ---- hoist loop-invariant small operands into registers
  // W_hh B-tiles: [gate q][kstep]  (128 VGPRs, resident across whole recurrence)
  v16h btile[4][4];
  #pragma unroll
  for(int q=0; q<4; ++q)
    #pragma unroll
    for(int ks=0; ks<4; ++ks)
      btile[q][ks] = *(const v16h*)&whhF[(((wv*4+q)*4+ks)*32 + lane)*16];
  // fc1 B-tiles (32 VGPRs)
  v16h f1tile[4];
  #pragma unroll
  for(int ks=0; ks<4; ++ks)
    f1tile[ks] = *(const v16h*)&fc1F[((wv*4+ks)*32 + lane)*16];
  // W_ih rows + combined bias for this lane's 4 gate columns (gate-permuted)
  float wihr[4][5], bsum[4];
  #pragma unroll
  for(int q=0; q<4; ++q){
    int ncolg = q*128 + wv*16 + m;         // original gate row index
    bsum[q] = bih[ncolg] + bhh[ncolg];
    #pragma unroll
    for(int k=0; k<5; ++k) wihr[q][k] = Wih[ncolg*5 + k];
  }
  // fc2 weights for this thread's reduction slice
  float f2w[8];
  {
    int p = tid & 15;
    #pragma unroll
    for(int j=0; j<8; ++j) f2w[j] = fc2w[p*8 + j];
  }
  const float fc1bias = fc1b[wv*16 + m];
  const float zw      = zonew[0];
  const float fc2bias = fc2b[0];

  // ---- init state: h = c = 1, TOut[:, :8] = T0, E = T0[:, 8]
  float creg[8];
  #pragma unroll
  for(int r=0; r<8; ++r) creg[r] = 1.0f;
  for(int e=tid; e<4*32*16; e+=256) sHf[0][e] = (_Float16)1.0f;
  for(int e=tid; e<16*WIN_; e+=256){
    int b = e>>3, t = e&7;
    sTout[b*T_ + t] = X[(rowBase+b)*(T_*7) + t*7 + 0];
  }
  if(tid < 16) sE[tid] = X[(rowBase+tid)*(T_*7) + WIN_*7 + 0];
  __syncthreads();

  for(int i=WIN_; i<T_; ++i){
    const bool  enc   = (i < ENCO_);
    const float ratio = (float)i / (float)ENCO_;

    // TOut[:, i] = E
    if(tid < 16) sTout[tid*T_ + i] = sE[tid];
    __syncthreads();

    // build embed window sX[b][t][k], window = [i-7, i]
    const int s0 = i - WIN_ + 1;
    for(int e=tid; e<16*WIN_*5; e+=256){
      int b = e/40, r = e%40, t = r/5, k = r%5;
      int tt = s0 + t;
      int row = rowBase + b;
      float v;
      if(k == 0){
        float to = sTout[b*T_ + tt];
        if(enc){
          float t0 = X[row*(T_*7) + tt*7 + 0];
          v = fmaf(t0, ratio, to*(1.0f-ratio));
        } else v = to;
      } else {
        v = X[row*(T_*7) + tt*7 + k];      // Ext_X channels 1..4
      }
      sX[e] = v;
    }
    __syncthreads();

    // ---- 8 sequential LSTM cell steps (1 barrier each, gates stay in registers)
    for(int t=0; t<WIN_; ++t){
      const _Float16* rb = sHf[t & 1];           // read h fragments
      _Float16*       wb = sHf[(t + 1) & 1];     // write next h fragments

      // preload A fragments first: ds loads overlap the C-init FMAs below
      v16h afr[4];
      #pragma unroll
      for(int ks=0; ks<4; ++ks)
        afr[ks] = *(const v16h*)&rb[ks*512 + lane*16];

      // C-init = x_t @ W_ih^T + b_ih + b_hh (gate-permuted C layout)
      v8f acc[4];
      #pragma unroll
      for(int q=0; q<4; ++q){
        #pragma unroll
        for(int r=0; r<8; ++r){
          const float* xr = &sX[(r+hi8)*40 + t*5];
          float c0 = bsum[q];
          c0 = fmaf(xr[0], wihr[q][0], c0);
          c0 = fmaf(xr[1], wihr[q][1], c0);
          c0 = fmaf(xr[2], wihr[q][2], c0);
          c0 = fmaf(xr[3], wihr[q][3], c0);
          c0 = fmaf(xr[4], wihr[q][4], c0);
          acc[q][r] = c0;
        }
      }
      // gate GEMM: h(16x128) @ W_hh^T -> 4 Ksteps x 4 gate-tiles of WMMA
      #pragma unroll
      for(int ks=0; ks<4; ++ks)
        #pragma unroll
        for(int q=0; q<4; ++q)
          acc[q] = __builtin_amdgcn_wmma_f32_16x16x32_f16(
              false, afr[ks], false, btile[q][ks], (short)0, acc[q], false, false);

      // cell update entirely in registers: acc[0..3][r] = gi,gf,gg,go for the
      // SAME hidden unit j = wv*16+m, row r+hi8 (thanks to gate permutation).
      // Branch-free activations -> straight-line TRANS/VALU code.
      #pragma unroll
      for(int r=0; r<8; ++r){
        float gi = acc[0][r], gf = acc[1][r], gg = acc[2][r], go = acc[3][r];
        float c = creg[r];
        c = sigf(gf)*c + sigf(gi)*tanhfast(gg);
        creg[r] = c;
        float h = sigf(go)*tanhfast(c);
        // scatter h (f16) into A-fragment order for next step
        int row = r + hi8;
        int j  = wv*16 + m;
        int ks = j >> 5, kk = j & 31;
        int ln = row + ((kk >> 4) << 4);
        int el = kk & 15;
        wb[ks*512 + ln*16 + el] = (_Float16)h;
      }
      __syncthreads();
    }

    // ---- fc1: o = relu(h @ fc1_w^T + b); wave wv owns N-tile wv
    // final h is in sHf[0] (8 steps -> parity back to buffer 0)
    {
      v8f acc = {};
      #pragma unroll
      for(int ks=0; ks<4; ++ks){
        v16h a = *(const v16h*)&sHf[0][ks*512 + lane*16];
        acc = __builtin_amdgcn_wmma_f32_16x16x32_f16(
            false, a, false, f1tile[ks], (short)0, acc, false, false);
      }
      int ncol = wv*16 + m;
      #pragma unroll
      for(int r=0; r<8; ++r)
        sO[(r+hi8)*HID_ + ncol] = fmaxf(acc[r] + fc1bias, 0.0f);
    }
    __syncthreads();

    // ---- fc2 (128 -> 1) tree reduction
    {
      int b = tid >> 4, p = tid & 15;
      float s = 0.f;
      #pragma unroll
      for(int j=0; j<8; ++j)
        s = fmaf(sO[b*HID_ + p*8 + j], f2w[j], s);
      sRed[tid] = s;
    }
    __syncthreads();
    if(tid < 16){
      float ext = fc2bias;
      #pragma unroll
      for(int p=0; p<16; ++p) ext += sRed[tid*16 + p];
      int row = rowBase + tid;
      float hv = X[row*(T_*7) + i*7 + 6];
      float it = intall[row*T_ + i];
      float total = ext + hv + it;
      float E = sE[tid];
      if(enc){
        float t0 = X[row*(T_*7) + i*7 + 0];
        E = ratio*t0 + (1.0f-ratio)*E + total*zw;
      } else {
        E = E + total*zw;
      }
      sE[tid] = E;
      out[2*BT_ + row*T_ + i] = ext;       // Ext_list
    }
    __syncthreads();
  }

  // ---- final outputs for this block's 16 rows
  for(int e=tid; e<16*T_; e+=256){
    int b = e >> 8, t = e & 255;
    int row = rowBase + b;
    out[row*T_ + t]          = sTout[b*T_ + t];                        // TOut
    out[BT_ + row*T_ + t]    = X[row*(T_*7) + t*7 + 6];                // HVAC_list
    if(t < WIN_) out[2*BT_ + row*T_ + t] = 0.0f;                       // Ext zeros
    out[3*BT_ + row*T_ + t]  = (t < WIN_) ? 0.0f : intall[row*T_ + t]; // Int_list
  }
}

// ---------------- host entry ----------------
extern "C" void kernel_launch(void* const* d_in, const int* in_sizes, int n_in,
                              void* d_out, int out_size, void* d_ws, size_t ws_size,
                              hipStream_t stream){
  const float* X    = (const float*)d_in[0];
  const float* Wih  = (const float*)d_in[1];
  const float* Whh  = (const float*)d_in[2];
  const float* bih  = (const float*)d_in[3];
  const float* bhh  = (const float*)d_in[4];
  const float* fc1w = (const float*)d_in[5];
  const float* fc1b = (const float*)d_in[6];
  const float* fc2w = (const float*)d_in[7];
  const float* fc2b = (const float*)d_in[8];
  const float* i1w  = (const float*)d_in[9];
  const float* i1b  = (const float*)d_in[10];
  const float* i3w  = (const float*)d_in[11];
  const float* i3b  = (const float*)d_in[12];
  const float* scw  = (const float*)d_in[13];
  const float* znw  = (const float*)d_in[14];
  float* out = (float*)d_out;

  char* ws = (char*)d_ws;
  _Float16* whhF = (_Float16*)ws;                       // 131072 B
  _Float16* fc1F = (_Float16*)(ws + 131072);            //  32768 B
  float*    ia   = (float*)(ws + 131072 + 32768);       // 524288 B

  pack_whh<<<(GATES_*HID_+255)/256, 256, 0, stream>>>(Whh, whhF);
  pack_fc1<<<(HID_*HID_+255)/256, 256, 0, stream>>>(fc1w, fc1F);
  int_all_kernel<<<(BT_+255)/256, 256, 0, stream>>>(X, i1w, i1b, i3w, i3b, scw, ia);
  modnn_main<<<B_/16, 256, 0, stream>>>(X, Wih, bih, bhh, fc1b, fc2w, fc2b, znw,
                                        whhF, fc1F, ia, out);
}
